// TernaryLinear_46084999086183
// MI455X (gfx1250) — matrix operand order repaired and checked by reading, not compiled, not run on hardware
//
#include <hip/hip_runtime.h>

// ---------------------------------------------------------------------------
// TernaryLinear on MI455X (gfx1250):
//   out = X(f32) @ ternary(W)^T + bias, all 4096x4096.
// Pre-pass: X -> f16, W -> ternary f16 into d_ws (64MB). Ternary weights are
// exact in f16; X rounds at ~2^-11. GEMM: v_wmma_f32_16x16x32_f16 with f32
// accumulation, async global->LDS double buffering.
// Tiling v2: block 128(M) x 256(N), 8 waves as 2x4, wave tile 64x64
//   -> per stage per wave: 16 ds_load_b128 : 16 wmma (1:1, was 1.5:1).
// ---------------------------------------------------------------------------

typedef _Float16 v8h  __attribute__((ext_vector_type(8)));
typedef _Float16 v16h __attribute__((ext_vector_type(16)));
typedef float    v4f  __attribute__((ext_vector_type(4)));
typedef float    v8f  __attribute__((ext_vector_type(8)));

#define N_FEAT 4096
#define BM 128
#define BN 256
#define BK 32
#define LDT 40           // padded LDS row stride in f16 (32 + 8) -> 16B aligned
#define NKSTEPS (N_FEAT / BK)

// ---------------- prep: X f32->f16, W f32->ternary f16 ----------------------
__global__ void prep_kernel(const float* __restrict__ x,
                            const float* __restrict__ w,
                            _Float16* __restrict__ xh,
                            _Float16* __restrict__ wh)
{
    const size_t groups = (size_t)N_FEAT * N_FEAT / 8;   // 8 elems per thread
    size_t gid = (size_t)blockIdx.x * blockDim.x + threadIdx.x;
    const bool isW = gid >= groups;
    const size_t i = (isW ? gid - groups : gid) * 8;
    const float* src = isW ? w : x;
    _Float16*    dst = isW ? wh : xh;

    v4f a = *(const v4f*)(src + i);
    v4f b = *(const v4f*)(src + i + 4);
    v8h o;
    if (isW) {
#pragma unroll
        for (int j = 0; j < 4; ++j) {
            float va = a[j], vb = b[j];
            o[j]     = (_Float16)((va > 0.5f) ? 1.0f : ((va < -0.5f) ? -1.0f : 0.0f));
            o[j + 4] = (_Float16)((vb > 0.5f) ? 1.0f : ((vb < -0.5f) ? -1.0f : 0.0f));
        }
    } else {
#pragma unroll
        for (int j = 0; j < 4; ++j) {
            o[j]     = (_Float16)a[j];
            o[j + 4] = (_Float16)b[j];
        }
    }
    *(v8h*)(dst + i) = o;
}

// ---------------- WMMA GEMM -------------------------------------------------

#define ASYNC_B128(ldsoff, base, goff)                                  \
    asm volatile("global_load_async_to_lds_b128 %0, %1, %2"             \
                 :: "v"(ldsoff), "v"(goff), "s"(base) : "memory")

__global__ __launch_bounds__(256)
void ternary_gemm_kernel(const _Float16* __restrict__ xh,
                         const _Float16* __restrict__ wh,
                         const float* __restrict__ bias,
                         float* __restrict__ out)
{
    __shared__ __align__(16) _Float16 As[2][BM * LDT];   // X tile [m][k]
    __shared__ __align__(16) _Float16 Bs[2][BN * LDT];   // W tile [n][k] (native)

    const int tid   = threadIdx.x;
    const int lane  = tid & 31;
    const int wave  = tid >> 5;
    const int waveM = wave & 1;              // 0..1 -> 64 rows each
    const int waveN = wave >> 1;             // 0..3 -> 64 cols each
    const int mBlock = blockIdx.y * BM;
    const int nBlock = blockIdx.x * BN;

    const unsigned asBase[2] = { (unsigned)(size_t)&As[0][0], (unsigned)(size_t)&As[1][0] };
    const unsigned bsBase[2] = { (unsigned)(size_t)&Bs[0][0], (unsigned)(size_t)&Bs[1][0] };

    // Per-stage async copy: A = 128 rows * 4 B128 = 512 chunks (2/thread),
    //                       B = 256 rows * 4 B128 = 1024 chunks (4/thread).
    // 6 async ops / thread / stage.
    auto issue_stage = [&](int buf, int kt) {
        const unsigned kOff = (unsigned)kt * BK;
#pragma unroll
        for (int j = 0; j < 2; ++j) {
            unsigned idx = (unsigned)tid * 2u + j;      // 0..511
            unsigned row = idx >> 2, kc = idx & 3;      // 4 B128 per row
            unsigned lds = asBase[buf] + (row * LDT + kc * 8) * 2;
            unsigned gof = ((unsigned)(mBlock + row) * N_FEAT + kOff + kc * 8) * 2;
            ASYNC_B128(lds, xh, gof);
        }
#pragma unroll
        for (int j = 0; j < 4; ++j) {
            unsigned idx = (unsigned)tid * 4u + j;      // 0..1023
            unsigned row = idx >> 2, kc = idx & 3;
            unsigned lds = bsBase[buf] + (row * LDT + kc * 8) * 2;
            unsigned gof = ((unsigned)(nBlock + row) * N_FEAT + kOff + kc * 8) * 2;
            ASYNC_B128(lds, wh, gof);
        }
    };

    v8f acc[4][4] = {};                      // 16 x 16x16 f32 accumulators

    // Fragment lane offsets (ISA VGPR layouts, CDNA5 ISA):
    //  A 16x32 f16: lane L holds row m=L%16; b128 #0 at k=(L>>4)*8, #1 at +16.
    //  B 32x16 f16: lane L holds col n=L%16; b128 #0 at k=(L>>4)*8, #1 at +16.
    const int rcLo = (lane & 15);
    const int aKLo = (lane >> 4) * 8;
    const int bKLo = (lane >> 4) * 8;

    issue_stage(0, 0);                        // prologue

    for (int kt = 0; kt < NKSTEPS; ++kt) {
        __syncthreads();                      // prev compute done before overwrite
        if (kt + 1 < NKSTEPS) {
            issue_stage((kt + 1) & 1, kt + 1);
            asm volatile("s_wait_asynccnt 0x6" ::: "memory");   // this stage landed
        } else {
            asm volatile("s_wait_asynccnt 0x0" ::: "memory");
        }
        __syncthreads();                      // all waves' loads visible

        const _Float16* A = &As[kt & 1][0];
        const _Float16* B = &Bs[kt & 1][0];

        v16h afr[4], bfr[4];
#pragma unroll
        for (int fm = 0; fm < 4; ++fm) {
            int off = (waveM * 64 + fm * 16 + rcLo) * LDT + aKLo;
            v8h lo = *(const v8h*)(A + off);
            v8h hi = *(const v8h*)(A + off + 16);
            afr[fm] = __builtin_shufflevector(lo, hi,
                        0,1,2,3,4,5,6,7,8,9,10,11,12,13,14,15);
        }
#pragma unroll
        for (int fn = 0; fn < 4; ++fn) {
            int off = (waveN * 64 + fn * 16 + rcLo) * LDT + bKLo;
            v8h lo = *(const v8h*)(B + off);
            v8h hi = *(const v8h*)(B + off + 16);
            bfr[fn] = __builtin_shufflevector(lo, hi,
                        0,1,2,3,4,5,6,7,8,9,10,11,12,13,14,15);
        }
#pragma unroll
        for (int fm = 0; fm < 4; ++fm)
#pragma unroll
            for (int fn = 0; fn < 4; ++fn)
                acc[fm][fn] = __builtin_amdgcn_wmma_f32_16x16x32_f16(
                    false, afr[fm], false, bfr[fn],
                    (short)0, acc[fm][fn], false, false);
    }

    // Epilogue: C/D layout — VGPR r: lanes 0-15 M=r, lanes 16-31 M=8+r; N=lane%16.
#pragma unroll
    for (int fm = 0; fm < 4; ++fm) {
#pragma unroll
        for (int fn = 0; fn < 4; ++fn) {
            int col = nBlock + waveN * 64 + fn * 16 + rcLo;
            float bv = bias[col];
            int rowBase = mBlock + waveM * 64 + fm * 16 + (lane >> 4) * 8;
#pragma unroll
            for (int r = 0; r < 8; ++r)
                out[(size_t)(rowBase + r) * N_FEAT + col] = acc[fm][fn][r] + bv;
        }
    }
}

// ---------------------------------------------------------------------------
extern "C" void kernel_launch(void* const* d_in, const int* in_sizes, int n_in,
                              void* d_out, int out_size, void* d_ws, size_t ws_size,
                              hipStream_t stream) {
    const float* x    = (const float*)d_in[0];
    const float* w    = (const float*)d_in[1];
    const float* bias = (const float*)d_in[2];
    float* out = (float*)d_out;

    // Workspace: 32MB f16 X + 32MB f16 ternary W (needs 64MB of d_ws).
    _Float16* xh = (_Float16*)d_ws;
    _Float16* wh = xh + (size_t)N_FEAT * N_FEAT;

    const int prepBlocks = (int)(2ull * N_FEAT * N_FEAT / 8 / 256);   // 16384
    prep_kernel<<<prepBlocks, 256, 0, stream>>>(x, w, xh, wh);

    dim3 grid(N_FEAT / BN, N_FEAT / BM);   // 16 x 32
    ternary_gemm_kernel<<<grid, 256, 0, stream>>>(xh, wh, bias, out);
}
